// QloraMLP_5669356831076
// MI455X (gfx1250) — matrix operand order repaired and checked by reading, not compiled, not run on hardware
//
#include <hip/hip_runtime.h>
#include <hip/hip_bf16.h>

// ---------------------------------------------------------------------------
// Types
// ---------------------------------------------------------------------------
typedef __attribute__((ext_vector_type(16))) __bf16 bf16x16;
typedef __attribute__((ext_vector_type(8)))  __bf16 bf16x8;
typedef __attribute__((ext_vector_type(8)))  float  f32x8;

__device__ __constant__ float kNF4[16] = {
    -1.0f, -0.6962f, -0.5251f, -0.3949f, -0.2844f, -0.1848f, -0.0911f, 0.0f,
     0.0796f, 0.1609f, 0.2461f, 0.3379f, 0.4407f, 0.5626f, 0.723f, 1.0f};

static constexpr float kNF4C[16] = {
    -1.0f, -0.6962f, -0.5251f, -0.3949f, -0.2844f, -0.1848f, -0.0911f, 0.0f,
     0.0796f, 0.1609f, 0.2461f, 0.3379f, 0.4407f, 0.5626f, 0.723f, 1.0f};

// ---------------------------------------------------------------------------
// Dequant pipeline (NF4 + double-quantized scalers)
// ---------------------------------------------------------------------------
__global__ __launch_bounds__(256) void k_block_absmax(
    const __bf16* __restrict__ w, float* __restrict__ scalers,
    float* __restrict__ partial, int nb) {
  int b = blockIdx.x * 256 + threadIdx.x;
  float m = 0.f;
  if (b < nb) {
    const __bf16* p = w + (size_t)b * 64;
#pragma unroll
    for (int i = 0; i < 8; ++i) {
      bf16x8 v = *(const bf16x8*)(p + i * 8);
#pragma unroll
      for (int j = 0; j < 8; ++j) m = fmaxf(m, fabsf((float)v[j]));
    }
    scalers[b] = m;
  }
  __shared__ float red[256];
  red[threadIdx.x] = m;
  __syncthreads();
#pragma unroll
  for (int s = 128; s > 0; s >>= 1) {
    if ((int)threadIdx.x < s) red[threadIdx.x] += red[threadIdx.x + s];
    __syncthreads();
  }
  if (threadIdx.x == 0) partial[blockIdx.x] = red[0];
}

__global__ __launch_bounds__(256) void k_mean(
    const float* __restrict__ partial, int nparts, int nb,
    float* __restrict__ meanOut) {
  __shared__ float red[256];
  float s = 0.f;
  for (int i = threadIdx.x; i < nparts; i += 256) s += partial[i];
  red[threadIdx.x] = s;
  __syncthreads();
#pragma unroll
  for (int st = 128; st > 0; st >>= 1) {
    if ((int)threadIdx.x < st) red[threadIdx.x] += red[threadIdx.x + st];
    __syncthreads();
  }
  if (threadIdx.x == 0) meanOut[0] = red[0] / (float)nb;
}

__global__ __launch_bounds__(256) void k_dq_scalers(
    const float* __restrict__ scalers, const float* __restrict__ meanPtr,
    float* __restrict__ deq) {
  int i = blockIdx.x * 256 + threadIdx.x;
  float mean = meanPtr[0];
  float c = scalers[i] - mean;
  __shared__ float red[256];
  red[threadIdx.x] = fabsf(c);
  __syncthreads();
#pragma unroll
  for (int s = 128; s > 0; s >>= 1) {
    if ((int)threadIdx.x < s)
      red[threadIdx.x] = fmaxf(red[threadIdx.x], red[threadIdx.x + s]);
    __syncthreads();
  }
  float amax = red[0];
  float qf = 256.0f / (2.0f * amax);
  float q = fminf(fmaxf(rintf(c * qf), -128.f), 127.f);
  deq[i] = q / qf + mean;
}

__global__ __launch_bounds__(256) void k_dequant(
    const __bf16* __restrict__ w, const float* __restrict__ scalers,
    const float* __restrict__ deq, __bf16* __restrict__ out, int n8) {
  __shared__ float nf4s[16];
  if (threadIdx.x < 16) nf4s[threadIdx.x] = kNF4[threadIdx.x];
  __syncthreads();
  int t = blockIdx.x * 256 + threadIdx.x;
  if (t >= n8) return;
  size_t e0 = (size_t)t * 8;
  int blk = t >> 3;
  float inv_s = 1.0f / scalers[blk];
  float d = deq[blk];
  bf16x8 v = *(const bf16x8*)(w + e0);
  bf16x8 o;
#pragma unroll
  for (int j = 0; j < 8; ++j) {
    float scaled = (float)v[j] * inv_s;
    int idx = 0;
#pragma unroll
    for (int k = 0; k < 15; ++k)
      idx += (scaled > 0.5f * (kNF4C[k] + kNF4C[k + 1])) ? 1 : 0;
    o[j] = (__bf16)(nf4s[idx] * d);
  }
  *(bf16x8*)(out + e0) = o;
}

// ---------------------------------------------------------------------------
// WMMA GEMM with async global->LDS pipeline:
//   C[M,N] = A[M,K] * B[N,K]^T   (bf16 in, f32 accumulate, bf16 out)
// FUSED variant also computes A*B1^T and applies silu(a)*b in the epilogue.
// Tile: 128(M) x 64(N), K-step 32, 8 waves as 4(M) x 2(N), each wave a
// 32x32 patch = 2x2 WMMA 16x16x32 subtiles. Double-buffered LDS filled by
// global_load_async_to_lds_b128 (ASYNCcnt), partial s_wait_asynccnt overlap.
// ---------------------------------------------------------------------------
#define BM 128
#define BN 64
#define BK 32
#define LP 40  // LDS row pitch in halves (32 data + 8 pad -> 80B, 16B aligned)

template <bool FUSED>
__global__ __launch_bounds__(256, 2) void gemm_wmma(
    const __bf16* __restrict__ A, const __bf16* __restrict__ B0,
    const __bf16* __restrict__ B1, __bf16* __restrict__ C, int Ndim, int K) {
  __shared__ __bf16 ldsA[2][BM * LP];
  __shared__ __bf16 ldsB[2][2 * BN * LP];

  const int tid  = threadIdx.x;
  const int lane = tid & 31;
  const int wave = tid >> 5;
  const int wm = wave & 3;   // wave M position (0..3)
  const int wn = wave >> 2;  // wave N position (0..1)

  const int tileM = blockIdx.x * BM;
  const int tileN = blockIdx.y * BN;

  const int nMats = FUSED ? 2 : 1;

  f32x8 acc[2][2][2];
#pragma unroll
  for (int i = 0; i < 2; ++i)
#pragma unroll
    for (int j = 0; j < 2; ++j)
#pragma unroll
      for (int m = 0; m < 2; ++m) acc[i][j][m] = (f32x8)0.0f;

  // Issue async global->LDS copies for one K-tile. Per wave:
  // A = 2 instructions, B = 2 (FUSED) or 1, so ASYNCcnt += 4 (or 3).
  auto issueAsync = [&](int k0, int buf) {
#pragma unroll
    for (int i = 0; i < 2; ++i) {
      int c = tid + i * 256;            // 0..511 16B chunks of the A tile
      int row = c >> 2, kc = c & 3;
      const __bf16* g = A + (size_t)(tileM + row) * K + k0 + kc * 8;
      unsigned l = (unsigned)(size_t)&ldsA[buf][row * LP + kc * 8];
      asm volatile("global_load_async_to_lds_b128 %0, %1, off"
                   :: "v"(l), "v"(g) : "memory");
    }
    if (FUSED) {
#pragma unroll
      for (int i = 0; i < 2; ++i) {
        int c = tid + i * 256;          // 0..511 chunks across both B tiles
        int mat = c >> 8, row = (c >> 2) & 63, kc = c & 3;
        const __bf16* Bp = mat ? B1 : B0;
        const __bf16* g = Bp + (size_t)(tileN + row) * K + k0 + kc * 8;
        unsigned l =
            (unsigned)(size_t)&ldsB[buf][mat * BN * LP + row * LP + kc * 8];
        asm volatile("global_load_async_to_lds_b128 %0, %1, off"
                     :: "v"(l), "v"(g) : "memory");
      }
    } else {
      int row = tid >> 2, kc = tid & 3;  // 256 chunks, one per thread
      const __bf16* g = B0 + (size_t)(tileN + row) * K + k0 + kc * 8;
      unsigned l = (unsigned)(size_t)&ldsB[buf][row * LP + kc * 8];
      asm volatile("global_load_async_to_lds_b128 %0, %1, off"
                   :: "v"(l), "v"(g) : "memory");
    }
    // speculative prefetch of the tile after next (global_prefetch_b8)
    if (k0 + BK < K) {
      __builtin_prefetch(A + (size_t)(tileM + (tid >> 2)) * K + k0 + BK, 0, 1);
      __builtin_prefetch(B0 + (size_t)(tileN + (tid & 63)) * K + k0 + BK, 0, 1);
    }
  };

  auto compute = [&](int buf) {
    // A fragment (16x32 bf16, CDNA5 layout):
    //   lane<16 : halves 0..7 = K 0..7,  halves 8..15 = K 16..23 (row M=lane)
    //   lane>=16: halves 0..7 = K 8..15, halves 8..15 = K 24..31 (row M=lane-16)
    bf16x16 af[2];
    const int mrow = wm * 32 + (lane & 15);
    const int alo = (lane < 16) ? 0 : 8;
#pragma unroll
    for (int sm = 0; sm < 2; ++sm) {
      const __bf16* base = &ldsA[buf][(mrow + sm * 16) * LP];
      bf16x8* p = (bf16x8*)&af[sm];
      p[0] = *(const bf16x8*)(base + alo);
      p[1] = *(const bf16x8*)(base + 16 + alo);
    }
    // B fragment (32x16 bf16): lane<16 holds K 0..7,16..23, lane>=16 holds K 8..15,24..31
    const int nrow = wn * 32 + (lane & 15);
    const int blo = (lane < 16) ? 0 : 8;
#pragma unroll
    for (int sn = 0; sn < 2; ++sn) {
#pragma unroll
      for (int mat = 0; mat < 2; ++mat) {
        if (mat >= nMats) break;
        const __bf16* base =
            &ldsB[buf][mat * BN * LP + (nrow + sn * 16) * LP + blo];
        bf16x16 bfrag;
        bf16x8* p = (bf16x8*)&bfrag;
        p[0] = *(const bf16x8*)(base);
        p[1] = *(const bf16x8*)(base + 16);
#pragma unroll
        for (int sm = 0; sm < 2; ++sm) {
          acc[sm][sn][mat] = __builtin_amdgcn_wmma_f32_16x16x32_bf16(
              false, af[sm], false, bfrag, (short)0, acc[sm][sn][mat], false,
              false);
        }
      }
    }
  };

  const int nIter = K / BK;
  issueAsync(0, 0);
  for (int it = 0; it < nIter; ++it) {
    int cur = it & 1;
    bool hasNext = (it + 1 < nIter);
    // Writes to buf cur^1 are safe: every wave passed the trailing barrier of
    // the previous iteration, which sealed all reads of that buffer.
    if (hasNext) issueAsync((it + 1) * BK, cur ^ 1);
    // Async loads complete in order: leaving the newest 4 (resp. 3)
    // outstanding means the current buffer's tile has fully landed in LDS.
    if (hasNext) {
      if (FUSED) asm volatile("s_wait_asynccnt 0x4" ::: "memory");
      else       asm volatile("s_wait_asynccnt 0x3" ::: "memory");
    } else {
      asm volatile("s_wait_asynccnt 0x0" ::: "memory");
    }
    __syncthreads();   // current buffer visible to all waves
    compute(cur);
    __syncthreads();   // all reads of current buffer done
  }

  // Epilogue. C/D layout: VGPR v, lanes 0-15 -> (M=v, N=lane),
  // lanes 16-31 -> (M=8+v, N=lane-16).
#pragma unroll
  for (int sm = 0; sm < 2; ++sm) {
#pragma unroll
    for (int sn = 0; sn < 2; ++sn) {
      int col  = tileN + wn * 32 + sn * 16 + (lane & 15);
      int row0 = tileM + wm * 32 + sm * 16 + ((lane < 16) ? 0 : 8);
#pragma unroll
      for (int v = 0; v < 8; ++v) {
        float o;
        if (FUSED) {
          float a = acc[sm][sn][0][v];
          float b = acc[sm][sn][1][v];
          o = (a / (1.0f + __expf(-a))) * b;  // silu(a) * b
        } else {
          o = acc[sm][sn][0][v];
        }
        C[(size_t)(row0 + v) * Ndim + col] = (__bf16)o;
      }
    }
  }
}

// ---------------------------------------------------------------------------
// Host-side orchestration
// ---------------------------------------------------------------------------
static void run_dequant(const __bf16* w, size_t n, float* scalers,
                        float* partial, float* meanBuf, float* deq,
                        __bf16* out, hipStream_t s) {
  int nb = (int)(n / 64);
  int ngrid = (nb + 255) / 256;
  k_block_absmax<<<ngrid, 256, 0, s>>>(w, scalers, partial, nb);
  k_mean<<<1, 256, 0, s>>>(partial, ngrid, nb, meanBuf);
  k_dq_scalers<<<nb / 256, 256, 0, s>>>(scalers, meanBuf, deq);
  int n8 = (int)(n / 8);
  k_dequant<<<(n8 + 255) / 256, 256, 0, s>>>(w, scalers, deq, out, n8);
}

extern "C" void kernel_launch(void* const* d_in, const int* in_sizes, int n_in,
                              void* d_out, int out_size, void* d_ws,
                              size_t ws_size, hipStream_t stream) {
  (void)n_in; (void)out_size; (void)ws_size;
  const __bf16* x  = (const __bf16*)d_in[0];
  const __bf16* w1 = (const __bf16*)d_in[1];
  const __bf16* w2 = (const __bf16*)d_in[2];
  const __bf16* w3 = (const __bf16*)d_in[3];

  const int d_model = 4096, hidden = 11008;
  const int M = in_sizes[0] / d_model;  // 8192
  const size_t nw = (size_t)d_model * hidden;
  const int nb = (int)(nw / 64);
  const int nparts = (nb + 255) / 256;

  char* ws = (char*)d_ws;
  __bf16* w1d = (__bf16*)ws; ws += nw * sizeof(__bf16);
  __bf16* w2d = (__bf16*)ws; ws += nw * sizeof(__bf16);
  __bf16* w3d = (__bf16*)ws; ws += nw * sizeof(__bf16);
  __bf16* h   = (__bf16*)ws; ws += (size_t)M * hidden * sizeof(__bf16);
  float* scalers = (float*)ws; ws += (size_t)nb * sizeof(float);
  float* deq     = (float*)ws; ws += (size_t)nb * sizeof(float);
  float* partial = (float*)ws; ws += (size_t)nparts * sizeof(float);
  float* meanBuf = (float*)ws; ws += 64;

  run_dequant(w1, nw, scalers, partial, meanBuf, deq, w1d, stream);
  run_dequant(w2, nw, scalers, partial, meanBuf, deq, w2d, stream);
  run_dequant(w3, nw, scalers, partial, meanBuf, deq, w3d, stream);

  // h = silu(x @ w1d^T) * (x @ w2d^T)   [M, hidden]
  dim3 g1(M / BM, hidden / BN);
  gemm_wmma<true><<<g1, 256, 0, stream>>>(x, w1d, w2d, h, hidden, d_model);

  // out = h @ w3d^T                      [M, d_model]
  dim3 g2(M / BM, d_model / BN);
  gemm_wmma<false><<<g2, 256, 0, stream>>>(h, w3d, nullptr, (__bf16*)d_out,
                                           d_model, hidden);
}
